// DeepseekMoE_73753178407351
// MI455X (gfx1250) — matrix operand order repaired and checked by reading, not compiled, not run on hardware
//
#include <hip/hip_runtime.h>
#include <hip/hip_bf16.h>

// Problem constants (from reference setup_inputs)
#define T_TOK   4096     // B*S = 2*2048 tokens
#define DDIM    1024
#define HDIM    512
#define NROUTED 8
#define NEXP    10       // 8 routed + 2 shared (gate forced to 1.0)
#define M_TILE  32       // tokens per block

typedef __attribute__((ext_vector_type(16))) __bf16 v16bf;
typedef __attribute__((ext_vector_type(8)))  float  v8f;
typedef __attribute__((ext_vector_type(8)))  unsigned short v8us;
typedef __attribute__((ext_vector_type(4)))  unsigned int   u32x4;
typedef __attribute__((ext_vector_type(4)))  int            i32x4;
typedef __attribute__((ext_vector_type(8)))  int            i32x8;

union Frag16 { v16bf v; v8us h[2]; };

__device__ __forceinline__ unsigned short f2bf(float f) {
  union { float f; unsigned u; } c; c.f = f;
  unsigned u = c.u;
  u += 0x7fffu + ((u >> 16) & 1u);        // round-to-nearest-even
  return (unsigned short)(u >> 16);
}

// A-matrix (16x32 bf16) fragment, ISA 7.12.2: lane half hl in {0,1};
// slots 0..7 -> K = hl*8+j, slots 8..15 -> K = 16+hl*8+j. `row` = 32 contiguous K.
__device__ __forceinline__ v16bf load_frag_a(const unsigned short* row, int hl) {
  Frag16 f;
  const unsigned short* p = row + hl * 8;
  f.h[0] = *(const v8us*)(p);
  f.h[1] = *(const v8us*)(p + 16);
  return f.v;
}
// B-matrix (32x16 bf16) fragment, same K order as A: slots 0..7 -> K = hl*8+j,
// slots 8..15 -> K = 16+hl*8+j; N-major weights give 2x b128 loads per lane.
__device__ __forceinline__ v16bf load_frag_b(const unsigned short* row, int hl) {
  Frag16 f;
  const unsigned short* p = row + hl * 8;
  f.h[0] = *(const v8us*)(p);
  f.h[1] = *(const v8us*)(p + 16);
  return f.v;
}

__device__ __forceinline__ v8f wmma_bf16(v16bf a, v16bf b, v8f c) {
  return __builtin_amdgcn_wmma_f32_16x16x32_bf16(false, a, false, b, (short)0, c,
                                                 false, false);
}

// Tensor Data Mover: 2D tile (tile_h rows of tile_w bf16 elems, row stride
// row_stride elems) from global -> LDS offset lds_off. D# packing per
// CDNA5 ISA (data_size=1 => 2 bytes, type=2).
__device__ __forceinline__ void tdm_load_2d_to_lds(unsigned lds_off,
                                                   const void* gptr,
                                                   unsigned tile_w, unsigned tile_h,
                                                   unsigned row_stride) {
  unsigned long long ga = (unsigned long long)(size_t)gptr;
  u32x4 g0;
  g0[0] = 1u;                                             // count=1, user desc
  g0[1] = lds_off;                                        // lds_addr (bytes)
  g0[2] = (unsigned)ga;                                   // global_addr[31:0]
  g0[3] = (unsigned)((ga >> 32) & 0x1FFFFFFu) | (2u << 30); // addr[56:32] | type=2
  i32x8 g1;
  g1[0] = (int)(1u << 16);                                // wg_mask=0, data_size=2B
  g1[1] = (int)((tile_w & 0xFFFFu) << 16);                // tensor_dim0[15:0]
  g1[2] = (int)((tile_w >> 16) | ((tile_h & 0xFFFFu) << 16)); // dim0 hi | dim1 lo
  g1[3] = (int)((tile_h >> 16) | ((tile_w & 0xFFFFu) << 16)); // dim1 hi | tile_dim0
  g1[4] = (int)(tile_h & 0xFFFFu);                        // tile_dim1 (tile_dim2=0)
  g1[5] = (int)row_stride;                                // tensor_dim0_stride lo32
  g1[6] = 0;
  g1[7] = 0;
  i32x4 gz = {0, 0, 0, 0};
#if defined(__clang_major__) && __clang_major__ >= 23
  i32x8 gz8 = {0, 0, 0, 0, 0, 0, 0, 0};
  __builtin_amdgcn_tensor_load_to_lds(g0, g1, gz, gz, gz8, 0);
#else
  __builtin_amdgcn_tensor_load_to_lds(g0, g1, gz, gz, 0);
#endif
}

__global__ __launch_bounds__(256) void moe_zero_kernel(float* __restrict__ p, int n) {
  for (int i = blockIdx.x * blockDim.x + threadIdx.x; i < n; i += gridDim.x * blockDim.x)
    p[i] = 0.f;
}

// One-time f32 -> bf16 elementwise convert (x).
__global__ __launch_bounds__(256) void cvt_bf16_kernel(const float* __restrict__ src,
                                                       unsigned short* __restrict__ dst,
                                                       int n) {
  for (int i = blockIdx.x * blockDim.x + threadIdx.x; i < n; i += gridDim.x * blockDim.x)
    dst[i] = f2bf(src[i]);
}

// One-time f32 [R][C] -> bf16 [C][R] transpose-convert (per expert z), LDS-tiled
// so both global sides stay coalesced. Output is N-major: B-fragment K runs
// contiguously in memory.
__global__ __launch_bounds__(256) void transpose_cvt_kernel(
    const float* __restrict__ src, unsigned short* __restrict__ dst, int R, int C) {
  __shared__ float tile[32][33];
  const int z = blockIdx.z;
  const float* s = src + (size_t)z * R * C;
  unsigned short* d = dst + (size_t)z * R * C;
  const int tx = threadIdx.x & 31;
  const int ty = threadIdx.x >> 5;          // 0..7
  const int c0 = blockIdx.x * 32;
  const int r0 = blockIdx.y * 32;
#pragma unroll
  for (int p = 0; p < 4; ++p)
    tile[ty + p * 8][tx] = s[(size_t)(r0 + ty + p * 8) * C + c0 + tx];
  __syncthreads();
#pragma unroll
  for (int p = 0; p < 4; ++p)
    d[(size_t)(c0 + ty + p * 8) * R + r0 + tx] = f2bf(tile[tx][ty + p * 8]);
}

// One wave32 per token: logits = x@router_w + b, top-2 softmax gates.
// gates[T][10]: slots 0..7 zero-masked routed gates, 8,9 = 1.0 (shared experts).
__global__ __launch_bounds__(256) void moe_router_kernel(
    const float* __restrict__ x, const float* __restrict__ rw,
    const float* __restrict__ rb, float* __restrict__ gates) {
  const int lane = threadIdx.x & 31;
  const int wv   = threadIdx.x >> 5;
  const int t    = blockIdx.x * 8 + wv;
  if (t >= T_TOK) return;

  float acc[NROUTED];
#pragma unroll
  for (int e = 0; e < NROUTED; ++e) acc[e] = 0.f;
  const float* xr = x + (size_t)t * DDIM;
  for (int d = lane; d < DDIM; d += 32) {
    float xv = xr[d];
#pragma unroll
    for (int e = 0; e < NROUTED; ++e) acc[e] += xv * rw[d * NROUTED + e];
  }
#pragma unroll
  for (int e = 0; e < NROUTED; ++e) {
#pragma unroll
    for (int off = 16; off > 0; off >>= 1)
      acc[e] += __shfl_xor(acc[e], off, 32);
  }
  if (lane == 0) {
    float lg[NROUTED];
#pragma unroll
    for (int e = 0; e < NROUTED; ++e) lg[e] = acc[e] + rb[e];
    int i1 = 0; float v1 = lg[0];
    for (int e = 1; e < NROUTED; ++e) if (lg[e] > v1) { v1 = lg[e]; i1 = e; }
    int i2 = -1; float v2 = -3.4e38f;
    for (int e = 0; e < NROUTED; ++e) if (e != i1 && lg[e] > v2) { v2 = lg[e]; i2 = e; }
    float e2 = __expf(v2 - v1);
    float g1 = 1.f / (1.f + e2);
    float g2 = e2 / (1.f + e2);
    float* gr = gates + (size_t)t * NEXP;
#pragma unroll
    for (int e = 0; e < NROUTED; ++e) gr[e] = (e == i1) ? g1 : (e == i2) ? g2 : 0.f;
    gr[8] = 1.f;
    gr[9] = 1.f;
  }
}

// Fused SwiGLU expert, all-bf16 datapath. Block = 32 tokens x 1 expert, 8 waves.
// A tiles: TDM DMA into LDS. B tiles: direct global b128 loads of N-major bf16
// weights (L2 resident). h stays in LDS. 8 barriers executed (6 in source).
__global__ __launch_bounds__(256) void moe_expert_kernel(
    const unsigned short* __restrict__ x_bf, const float* __restrict__ gates,
    const unsigned short* __restrict__ w1t, const unsigned short* __restrict__ w3t,
    const unsigned short* __restrict__ w2t, float* __restrict__ out) {
  // 64KB LDS: [0,32KB) x half-K tile (also gate cache in GEMM2),
  //           [32KB,64KB) h (also skip-flag before GEMM1 writes it).
  __shared__ unsigned short lds_all[32768];
  unsigned short* x_sh = lds_all;           // LDS byte offset 0 (single alloc)
  unsigned short* h_sh = lds_all + 16384;

  const int tid  = threadIdx.x;
  const int lane = tid & 31;
  const int wv   = tid >> 5;
  const int hl   = lane >> 4;
  const int ln   = lane & 15;
  const int tok0 = blockIdx.x * M_TILE;
  const int e    = blockIdx.y;

  int* flag = (int*)h_sh;
  if (tid == 0) *flag = 0;
  __syncthreads();
  if (tid < M_TILE && gates[(size_t)(tok0 + tid) * NEXP + e] != 0.f)
    atomicOr(flag, 1);
  __syncthreads();
  if (*flag == 0) return;     // no token in this tile routed to expert e

  const unsigned short* w1e = w1t + (size_t)e * HDIM * DDIM;  // [H][D] bf16
  const unsigned short* w3e = w3t + (size_t)e * HDIM * DDIM;  // [H][D] bf16
  const unsigned short* w2e = w2t + (size_t)e * DDIM * HDIM;  // [D][H] bf16

  const v8f vz = {0.f, 0.f, 0.f, 0.f, 0.f, 0.f, 0.f, 0.f};

  // ---------------- GEMM1: h1 = x@w1, h3 = x@w3. Wave owns 64 H-cols (4 tiles).
  v8f acc1[2][4], acc3[2][4];
#pragma unroll
  for (int mr = 0; mr < 2; ++mr)
#pragma unroll
    for (int ct = 0; ct < 4; ++ct) { acc1[mr][ct] = vz; acc3[mr][ct] = vz; }

  for (int kh = 0; kh < 2; ++kh) {          // K = 1024 in two 512 halves
    __syncthreads();                        // x_sh consumers done
    if (tid < 32) {                         // wave 0 drives the Tensor Data Mover
      tdm_load_2d_to_lds(0u, x_bf + (size_t)tok0 * DDIM + kh * 512,
                         512u, (unsigned)M_TILE, (unsigned)DDIM);
      __builtin_amdgcn_s_wait_tensorcnt(0);
    }
    __syncthreads();                        // tile visible to all 8 waves
    for (int kk = 0; kk < 512; kk += 32) {
      const int k0 = kh * 512 + kk;
      v16bf a[2];
#pragma unroll
      for (int mr = 0; mr < 2; ++mr)
        a[mr] = load_frag_a(&x_sh[(mr * 16 + ln) * 512 + kk], hl);
#pragma unroll
      for (int ct = 0; ct < 4; ++ct) {
        const int n = wv * 64 + ct * 16 + ln;
        v16bf b1 = load_frag_b(&w1e[(size_t)n * DDIM + k0], hl);
        v16bf b3 = load_frag_b(&w3e[(size_t)n * DDIM + k0], hl);
#pragma unroll
        for (int mr = 0; mr < 2; ++mr) {
          acc1[mr][ct] = wmma_bf16(a[mr], b1, acc1[mr][ct]);
          acc3[mr][ct] = wmma_bf16(a[mr], b3, acc3[mr][ct]);
        }
      }
      if (kk + 32 < 512)
        __builtin_prefetch(&w1e[(size_t)(wv * 64 + ln) * DDIM + k0 + 32], 0, 1);
    }
  }
  // SwiGLU epilogue -> h_sh bf16. C layout: VGPR r holds M=r(+8 for hi lanes).
#pragma unroll
  for (int mr = 0; mr < 2; ++mr)
#pragma unroll
    for (int ct = 0; ct < 4; ++ct)
#pragma unroll
      for (int r = 0; r < 8; ++r) {
        float h1v = acc1[mr][ct][r], h3v = acc3[mr][ct][r];
        float hv = (h1v / (1.f + __expf(-h1v))) * h3v;   // silu(h1)*h3
        int m = mr * 16 + r + 8 * hl;
        h_sh[m * HDIM + wv * 64 + ct * 16 + ln] = f2bf(hv);
      }

  __syncthreads();                          // h complete; x_sh free for gates
  float* gate_sh = (float*)x_sh;
  if (tid < M_TILE) gate_sh[tid] = gates[(size_t)(tok0 + tid) * NEXP + e];
  __syncthreads();

  // ---------------- GEMM2: out += gate * (h @ w2). Wave owns 128 D-cols (8 tiles).
  v8f acc[2][8];
#pragma unroll
  for (int mr = 0; mr < 2; ++mr)
#pragma unroll
    for (int ct = 0; ct < 8; ++ct) acc[mr][ct] = vz;

  for (int k0 = 0; k0 < HDIM; k0 += 32) {
    v16bf a[2];
#pragma unroll
    for (int mr = 0; mr < 2; ++mr)
      a[mr] = load_frag_a(&h_sh[(mr * 16 + ln) * HDIM + k0], hl);
#pragma unroll
    for (int ct = 0; ct < 8; ++ct) {
      const int n = wv * 128 + ct * 16 + ln;
      v16bf b = load_frag_b(&w2e[(size_t)n * HDIM + k0], hl);
#pragma unroll
      for (int mr = 0; mr < 2; ++mr)
        acc[mr][ct] = wmma_bf16(a[mr], b, acc[mr][ct]);
    }
  }
  // Gated accumulate into global output (f32 atomics across 10 expert planes).
#pragma unroll
  for (int mr = 0; mr < 2; ++mr)
#pragma unroll
    for (int ct = 0; ct < 8; ++ct)
#pragma unroll
      for (int r = 0; r < 8; ++r) {
        int m = mr * 16 + r + 8 * hl;
        float g = gate_sh[m];
        if (g != 0.f) {
          int col = wv * 128 + ct * 16 + ln;
          atomicAdd(&out[(size_t)(tok0 + m) * DDIM + col], g * acc[mr][ct][r]);
        }
      }
}

extern "C" void kernel_launch(void* const* d_in, const int* in_sizes, int n_in,
                              void* d_out, int out_size, void* d_ws, size_t ws_size,
                              hipStream_t stream) {
  const float* x         = (const float*)d_in[0];
  const float* router_w  = (const float*)d_in[1];
  const float* router_b  = (const float*)d_in[2];
  const float* routed_w1 = (const float*)d_in[3];
  const float* routed_w3 = (const float*)d_in[4];
  const float* routed_w2 = (const float*)d_in[5];
  const float* shared_w1 = (const float*)d_in[6];
  const float* shared_w3 = (const float*)d_in[7];
  const float* shared_w2 = (const float*)d_in[8];
  float* out = (float*)d_out;

  // Workspace layout (~40 MB): bf16 x, N-major bf16 weights (L2-resident), gates.
  unsigned short* x_bf = (unsigned short*)d_ws;                  // [T][D]
  unsigned short* w1t  = x_bf + (size_t)T_TOK * DDIM;            // [10][H][D]
  unsigned short* w3t  = w1t + (size_t)NEXP * HDIM * DDIM;       // [10][H][D]
  unsigned short* w2t  = w3t + (size_t)NEXP * HDIM * DDIM;       // [10][D][H]
  float* gates = (float*)(w2t + (size_t)NEXP * DDIM * HDIM);     // [T][10]

  moe_zero_kernel<<<2048, 256, 0, stream>>>(out, out_size);
  cvt_bf16_kernel<<<2048, 256, 0, stream>>>(x, x_bf, T_TOK * DDIM);

  dim3 gw1(HDIM / 32, DDIM / 32, NROUTED);   // src [D][H] -> dst [H][D]
  transpose_cvt_kernel<<<gw1, 256, 0, stream>>>(routed_w1, w1t, DDIM, HDIM);
  transpose_cvt_kernel<<<gw1, 256, 0, stream>>>(routed_w3, w3t, DDIM, HDIM);
  dim3 gw1s(HDIM / 32, DDIM / 32, 2);
  transpose_cvt_kernel<<<gw1s, 256, 0, stream>>>(shared_w1,
      w1t + (size_t)NROUTED * HDIM * DDIM, DDIM, HDIM);
  transpose_cvt_kernel<<<gw1s, 256, 0, stream>>>(shared_w3,
      w3t + (size_t)NROUTED * HDIM * DDIM, DDIM, HDIM);
  dim3 gw2(DDIM / 32, HDIM / 32, NROUTED);   // src [H][D] -> dst [D][H]
  transpose_cvt_kernel<<<gw2, 256, 0, stream>>>(routed_w2, w2t, HDIM, DDIM);
  dim3 gw2s(DDIM / 32, HDIM / 32, 2);
  transpose_cvt_kernel<<<gw2s, 256, 0, stream>>>(shared_w2,
      w2t + (size_t)NROUTED * DDIM * HDIM, HDIM, DDIM);

  moe_router_kernel<<<T_TOK / 8, 256, 0, stream>>>(x, router_w, router_b, gates);

  dim3 ge(T_TOK / M_TILE, NEXP);
  moe_expert_kernel<<<ge, 256, 0, stream>>>(x_bf, gates, w1t, w3t, w2t, out);
}
